// SparseAttention_88038239633975
// MI455X (gfx1250) — matrix operand order repaired and checked by reading, not compiled, not run on hardware
//
#include <hip/hip_runtime.h>
#include <hip/hip_bf16.h>

// ---------------------------------------------------------------------------
// Sparse (BigBird-style) attention for MI455X / gfx1250.
// All matmuls on v_wmma_f32_16x16x32_bf16 (fp32 -> bf16 operands, f32 accum).
// GEMM weight tiles staged into LDS by the Tensor Data Mover (TDM), double
// buffered, gated with s_wait_tensorcnt + workgroup barrier.
// ---------------------------------------------------------------------------

typedef __bf16 bf16;
typedef __attribute__((ext_vector_type(8)))  __bf16 v8bf;
typedef __attribute__((ext_vector_type(16))) __bf16 v16bf;
typedef __attribute__((ext_vector_type(8)))  float  v8f;
typedef __attribute__((ext_vector_type(4)))  unsigned int u32x4;
typedef __attribute__((ext_vector_type(8)))  int    i32x8;
typedef __attribute__((ext_vector_type(4)))  int    i32x4;

#define HIDDEN      2048
#define HEADS       16
#define HEAD_DIM    128
#define QBLOCK      64
#define SEQ         4096
#define NUM_BLOCKS  64
#define ATT_SCALE   0.08838834764831845f   // 1/sqrt(128)

// ---------------------------------------------------------------------------
// Fragment helpers (wave32; A/B use this code's contiguous K order, C/D per ISA 7.12.2)
// A 16x32 bf16 : row M = lane&15; lane<16 holds K 0-15 contiguous, lane>=16 K 16-31 contiguous
// B 32x16 bf16 : column N = lane&15; lane<16 holds K 0-15 contiguous,
//                lane>=16 holds K 16-31 contiguous
// C/D 16x16 f32: VGPR v -> M=v (lanes 0-15) / M=v+8 (lanes 16-31), N=lane&15
// ---------------------------------------------------------------------------
__device__ __forceinline__ v16bf join16(v8bf lo, v8bf hi) {
    return __builtin_shufflevector(lo, hi,
        0,1,2,3,4,5,6,7,8,9,10,11,12,13,14,15);
}

__device__ __forceinline__ v16bf load_a_frag(const bf16* row, int k0, int hiHalf) {
    const bf16* p = row + k0 + hiHalf * 16;         // 16B aligned
    return join16(*(const v8bf*)p, *(const v8bf*)(p + 8));
}

__device__ __forceinline__ v16bf load_b_frag(const bf16* colRow, int k0, int hiHalf) {
    const bf16* p = colRow + k0 + hiHalf * 16;      // 16B aligned, 16 contiguous K
    return join16(*(const v8bf*)p, *(const v8bf*)(p + 8));
}

__device__ __forceinline__ v8f wmma_bf16(v16bf a, v16bf b, v8f c) {
    return __builtin_amdgcn_wmma_f32_16x16x32_bf16(
        /*neg_a=*/false, a, /*neg_b=*/false, b,
        /*c_mod=*/(short)0, c, /*reuse_a=*/false, /*reuse_b=*/false);
}

// ---------------------------------------------------------------------------
// TDM: DMA a 2D bf16 tile [tileRows x tileK] (row stride = strideElems) from
// global memory into LDS, inserting 4 DWORDs of padding after every 16 DWORDs
// stored (so a 32-elem (64B) row occupies 80B in LDS -> conflict-free
// ds_load_b128 column fetches: 80B = 20 banks, gcd(20,64)=4).
// D# layout per CDNA5 ISA sec.8 (groups 0..3).
// This toolchain exposes the 6-arg builtin (extra int32x8 group before cpol).
// ---------------------------------------------------------------------------
__device__ __forceinline__ void tdm_load_tile_2d(unsigned int ldsByteOff,
                                                 const void* gaddr,
                                                 int tileK, int tileRows,
                                                 int strideElems) {
    const unsigned long long ga = (unsigned long long)gaddr;
    u32x4 g0;
    g0[0] = 1u;                                        // count=1, user mode
    g0[1] = ldsByteOff;                                // lds_addr
    g0[2] = (unsigned int)(ga & 0xFFFFFFFFu);          // global_addr[31:0]
    g0[3] = (unsigned int)((ga >> 32) & 0x1FFFFFFu)    // global_addr[56:32]
          | (2u << 30);                                // type=2 ("image")
    const unsigned int tdim0 = 1u << 20;               // generous OOB bounds
    const unsigned int tdim1 = 1u << 20;
    i32x8 g1;
    g1[0] = (int)((1u << 16)                           // data_size = 2 bytes
          |       (1u << 20)                           // pad_enable
          |       (3u << 22)                           // pad_interval = 16 DW
          |       (3u << 25));                         // pad_amount   = 4 DW
    g1[1] = (int)((tdim0 & 0xFFFFu) << 16);            // tensor_dim0[15:0]
    g1[2] = (int)((tdim0 >> 16) | ((tdim1 & 0xFFFFu) << 16));
    g1[3] = (int)((tdim1 >> 16) | ((unsigned int)tileK << 16));  // tile_dim0
    g1[4] = tileRows;                                  // tile_dim1 (tile_dim2=0)
    g1[5] = strideElems;                               // tensor_dim0_stride lo32
    g1[6] = 0;
    g1[7] = 0;
    i32x4 z4 = {0, 0, 0, 0};                           // groups 2/3: 2D tensor
    i32x8 z8 = {0, 0, 0, 0, 0, 0, 0, 0};
    __builtin_amdgcn_tensor_load_to_lds(g0, g1, z4, z4, z8, 0);
}

// ---------------------------------------------------------------------------
// fp32 -> bf16 conversion (grid-stride)
// ---------------------------------------------------------------------------
__global__ void cvt_f32_bf16(const float* __restrict__ in,
                             bf16* __restrict__ out, int n) {
    int i = blockIdx.x * blockDim.x + threadIdx.x;
    int stride = gridDim.x * blockDim.x;
    for (; i < n; i += stride) out[i] = (bf16)in[i];
}

// ---------------------------------------------------------------------------
// Y[M,N] = A[M,K] @ W[N,K]^T   (torch Linear), bf16 operands, f32 accumulate.
// block = 256 threads (8 waves), block tile 128(M) x 256(N),
// wave tile 64x64 = 4x4 WMMA accumulators.
// W tile (256 rows x 32 K, 16KB) staged to LDS by TDM, double buffered;
// A fragments stream from global (L2-resident).
// ---------------------------------------------------------------------------
#define B_ROW_LDS   80                      // 64B data + 16B TDM pad
#define B_BUF_BYTES (256 * B_ROW_LDS)       // 20 KB per buffer

template<bool OUT_BF16>
__global__ __launch_bounds__(256) void gemm_bf16(
        const bf16* __restrict__ A, const bf16* __restrict__ W,
        void* __restrict__ Cout, int M, int N, int K) {
    __shared__ __align__(16) char sB[2 * B_BUF_BYTES];

    const int lane   = threadIdx.x & 31;
    const int wave   = threadIdx.x >> 5;
    const int hiHalf = lane >> 4;
    const int lr     = lane & 15;
    const int mBase  = blockIdx.x * 128 + (wave & 1) * 64;
    const int nWave  = wave >> 1;                       // 0..3
    const int nBase  = blockIdx.y * 256 + nWave * 64;

    v8f acc[4][4];
    #pragma unroll
    for (int i = 0; i < 4; i++)
        #pragma unroll
        for (int j = 0; j < 4; j++)
            acc[i][j] = (v8f){0.f,0.f,0.f,0.f,0.f,0.f,0.f,0.f};

    const bf16* aRow0     = A + (size_t)(mBase + lr) * K;
    const bf16* wTileBase = W + (size_t)(blockIdx.y * 256) * K;
    const int   nSteps    = K / 32;

    // prologue: DMA first W tile
    if (wave == 0)
        tdm_load_tile_2d((unsigned int)(size_t)&sB[0], wTileBase, 32, 256, K);

    for (int kk = 0; kk < nSteps; ++kk) {
        if (wave == 0) __builtin_amdgcn_s_wait_tensorcnt(0);
        __syncthreads();   // tile kk ready; also fences reuse of buf[(kk+1)&1]
        if (wave == 0 && kk + 1 < nSteps)
            tdm_load_tile_2d((unsigned int)(size_t)&sB[((kk + 1) & 1) * B_BUF_BYTES],
                             (const char*)wTileBase + (size_t)(kk + 1) * 32 * 2,
                             32, 256, K);

        const char* bBuf = &sB[(kk & 1) * B_BUF_BYTES];
        const int   k0   = kk * 32;

        v16bf a[4];
        #pragma unroll
        for (int mt = 0; mt < 4; mt++)
            a[mt] = load_a_frag(aRow0 + (size_t)mt * 16 * K, k0, hiHalf);
        #pragma unroll
        for (int nt = 0; nt < 4; nt++) {
            const char* bp = bBuf + (nWave * 64 + nt * 16 + lr) * B_ROW_LDS
                                  + hiHalf * 32;
            v16bf b = join16(*(const v8bf*)bp, *(const v8bf*)(bp + 16));
            #pragma unroll
            for (int mt = 0; mt < 4; mt++)
                acc[mt][nt] = wmma_bf16(a[mt], b, acc[mt][nt]);
        }
    }

    #pragma unroll
    for (int mt = 0; mt < 4; mt++) {
        #pragma unroll
        for (int nt = 0; nt < 4; nt++) {
            const int col = nBase + nt * 16 + lr;
            #pragma unroll
            for (int v = 0; v < 8; v++) {
                const int row = mBase + mt * 16 + v + hiHalf * 8;
                const float val = acc[mt][nt][v];
                if constexpr (OUT_BF16)
                    ((bf16*)Cout)[(size_t)row * N + col] = (bf16)val;
                else
                    ((float*)Cout)[(size_t)row * N + col] = val;
            }
        }
    }
}

// ---------------------------------------------------------------------------
// Sparse block attention. One 128-thread (4-wave) WG per (query block, head).
// Wave w owns query rows [16w, 16w+16). All selected key blocks are fully
// unmasked (mask is block-granular), so no -inf handling is needed.
// LDS: scores[64][388] f32 (reused in place as bf16 probs) + Vt[128][72] bf16
// ---------------------------------------------------------------------------
#define SCORE_STRIDE 388                         // floats per score row
#define VT_STRIDE    72                          // bf16 per Vt row (16B-aligned)
#define SMEM_SCORES  (64 * SCORE_STRIDE * 4)     // 99328 B
#define SMEM_VT      (128 * VT_STRIDE * 2)       // 18432 B

__global__ __launch_bounds__(128) void sparse_attn(
        const bf16* __restrict__ Q, const bf16* __restrict__ Kg,
        const bf16* __restrict__ V, const int* __restrict__ rand_idx,
        bf16* __restrict__ O) {
    __shared__ __align__(16) char smem[SMEM_SCORES + SMEM_VT];
    float* sS  = (float*)smem;                       // [64][SCORE_STRIDE]
    bf16*  sVt = (bf16*)(smem + SMEM_SCORES);        // [128][VT_STRIDE]

    const int qb   = blockIdx.x;
    const int h    = blockIdx.y;
    const int tid  = threadIdx.x;
    const int lane = tid & 31;
    const int wave = tid >> 5;
    const int hiHalf = lane >> 4;
    const int lr     = lane & 15;

    // --- selected key blocks: global{0, NB-1} + local{qb-1,qb,qb+1} + random
    int cand[6] = {0, NUM_BLOCKS - 1, qb - 1, qb, qb + 1, rand_idx[qb]};
    int sel[6]; int nsel = 0;
    for (int c = 0; c < 6; c++) {
        int b = cand[c];
        if (b < 0 || b >= NUM_BLOCKS) continue;
        bool dup = false;
        for (int s = 0; s < nsel; s++) dup |= (sel[s] == b);
        if (!dup) sel[nsel++] = b;
    }

    // --- Q A-fragments for this wave's 16 rows (K = head_dim = 128 -> 4 steps)
    const bf16* qRow =
        Q + (size_t)(qb * QBLOCK + wave * 16 + lr) * HIDDEN + h * HEAD_DIM;
    v16bf aQ[4];
    #pragma unroll
    for (int ks = 0; ks < 4; ks++) aQ[ks] = load_a_frag(qRow, ks * 32, hiHalf);

    // --- phase 1: S = (Q K^T) * scale -> LDS
    // B fragment of K^T: column n = key row -> contiguous head-dim of K row.
    for (int b = 0; b < nsel; b++) {
        const int kb = sel[b];
        const bf16* kBase = Kg + (size_t)(kb * QBLOCK) * HIDDEN + h * HEAD_DIM;
        #pragma unroll
        for (int nt = 0; nt < 4; nt++) {
            v8f acc = (v8f){0.f,0.f,0.f,0.f,0.f,0.f,0.f,0.f};
            const bf16* kc = kBase + (size_t)(nt * 16 + lr) * HIDDEN;
            #pragma unroll
            for (int ks = 0; ks < 4; ks++)
                acc = wmma_bf16(aQ[ks], load_b_frag(kc, ks * 32, hiHalf), acc);
            #pragma unroll
            for (int v = 0; v < 8; v++) {
                const int m = wave * 16 + v + hiHalf * 8;
                sS[m * SCORE_STRIDE + b * 64 + nt * 16 + lr] = acc[v] * ATT_SCALE;
            }
        }
    }
    __syncthreads();

    // --- phase 2: row softmax; rewrite probs as bf16 in place
    // (bf16 write at byte 2c always trails the f32 read at byte 4c of the
    //  same row, and each row is owned by one thread -> no hazard)
    if (tid < 64) {
        const int ncol = nsel * 64;
        float* row = sS + (size_t)tid * SCORE_STRIDE;
        float mx = -1e30f;
        for (int c = 0; c < ncol; c++) mx = fmaxf(mx, row[c]);
        float sum = 0.f;
        for (int c = 0; c < ncol; c++) { float e = __expf(row[c] - mx); row[c] = e; sum += e; }
        const float inv = 1.f / sum;
        bf16* prow = (bf16*)row;
        for (int c = 0; c < ncol; c++) prow[c] = (bf16)(row[c] * inv);
    }
    __syncthreads();

    // --- phase 3: O = P V  (stage V transposed per block, then WMMA)
    v8f accO[8];
    #pragma unroll
    for (int nt = 0; nt < 8; nt++)
        accO[nt] = (v8f){0.f,0.f,0.f,0.f,0.f,0.f,0.f,0.f};

    for (int b = 0; b < nsel; b++) {
        const int kb = sel[b];
        // stage Vt[d][k]: 64 key rows x 128 dims, 8 dims per unit
        for (int it = tid; it < 64 * 16; it += 128) {
            const int kr = it >> 4;
            const int dg = it & 15;
            v8bf vv = *(const v8bf*)(V + (size_t)(kb * QBLOCK + kr) * HIDDEN
                                       + h * HEAD_DIM + dg * 8);
            #pragma unroll
            for (int j = 0; j < 8; j++)
                sVt[(dg * 8 + j) * VT_STRIDE + kr] = vv[j];
        }
        __syncthreads();

        const bf16* pRow = (const bf16*)(sS + (size_t)(wave * 16 + lr) * SCORE_STRIDE);
        #pragma unroll
        for (int ks2 = 0; ks2 < 2; ks2++) {
            v16bf aP = load_a_frag(pRow, b * 64 + ks2 * 32, hiHalf);
            #pragma unroll
            for (int nt = 0; nt < 8; nt++) {
                const bf16* vc = sVt + (size_t)(nt * 16 + lr) * VT_STRIDE;
                accO[nt] = wmma_bf16(aP, load_b_frag(vc, ks2 * 32, hiHalf), accO[nt]);
            }
        }
        __syncthreads();
    }

    // --- store O (head-interleaved layout [S, HIDDEN], head h at cols h*128..)
    #pragma unroll
    for (int nt = 0; nt < 8; nt++) {
        #pragma unroll
        for (int v = 0; v < 8; v++) {
            const int row = qb * QBLOCK + wave * 16 + v + hiHalf * 8;
            const int col = h * HEAD_DIM + nt * 16 + lr;
            O[(size_t)row * HIDDEN + col] = (bf16)accO[nt][v];
        }
    }
}

// ---------------------------------------------------------------------------
// host-side orchestration
// ws layout (bf16 elems): Xb | Wqb Wkb Wvb Wob | Qb Kb Vb | Ab  = 112 MB
// ---------------------------------------------------------------------------
extern "C" void kernel_launch(void* const* d_in, const int* in_sizes, int n_in,
                              void* d_out, int out_size, void* d_ws, size_t ws_size,
                              hipStream_t stream) {
    const float* X   = (const float*)d_in[0];
    const int*   rnd = (const int*)  d_in[1];
    const float* Wq  = (const float*)d_in[2];
    const float* Wk  = (const float*)d_in[3];
    const float* Wv  = (const float*)d_in[4];
    const float* Wo  = (const float*)d_in[5];
    float* out = (float*)d_out;

    const size_t XN = (size_t)SEQ * HIDDEN;       // 8M elems
    const size_t WN = (size_t)HIDDEN * HIDDEN;    // 4M elems
    bf16* Xb  = (bf16*)d_ws;
    bf16* Wqb = Xb  + XN;
    bf16* Wkb = Wqb + WN;
    bf16* Wvb = Wkb + WN;
    bf16* Wob = Wvb + WN;
    bf16* Qb  = Wob + WN;
    bf16* Kb  = Qb  + XN;
    bf16* Vb  = Kb  + XN;
    bf16* Ab  = Vb  + XN;

    cvt_f32_bf16<<<512, 256, 0, stream>>>(X,  Xb,  (int)XN);
    cvt_f32_bf16<<<512, 256, 0, stream>>>(Wq, Wqb, (int)WN);
    cvt_f32_bf16<<<512, 256, 0, stream>>>(Wk, Wkb, (int)WN);
    cvt_f32_bf16<<<512, 256, 0, stream>>>(Wv, Wvb, (int)WN);
    cvt_f32_bf16<<<512, 256, 0, stream>>>(Wo, Wob, (int)WN);

    dim3 gg(SEQ / 128, HIDDEN / 256);
    gemm_bf16<true><<<gg, 256, 0, stream>>>(Xb, Wqb, Qb, SEQ, HIDDEN, HIDDEN);
    gemm_bf16<true><<<gg, 256, 0, stream>>>(Xb, Wkb, Kb, SEQ, HIDDEN, HIDDEN);
    gemm_bf16<true><<<gg, 256, 0, stream>>>(Xb, Wvb, Vb, SEQ, HIDDEN, HIDDEN);

    sparse_attn<<<dim3(NUM_BLOCKS, HEADS), 128, 0, stream>>>(Qb, Kb, Vb, rnd, Ab);

    gemm_bf16<false><<<gg, 256, 0, stream>>>(Ab, Wob, (void*)out, SEQ, HIDDEN, HIDDEN);
}
